// PraxisAttention_62345745268775
// MI455X (gfx1250) — matrix operand order repaired and checked by reading, not compiled, not run on hardware
//
#include <hip/hip_runtime.h>
#include <hip/hip_bf16.h>
#include <math.h>

// ---------------------------------------------------------------------------
// PraxisAttention for MI455X (gfx1250): bf16 WMMA pipeline + TDM staging.
//   B=2 S=2048 DM=1024 H=16 HD=64, 10 memory tokens from top-k retrieval.
// ---------------------------------------------------------------------------

#define B_    2
#define S_    2048
#define DM_   1024
#define H_    16
#define HD_   64
#define MEMSZ 1000
#define KT    10
#define NEGBIG (-1e9f)

typedef __bf16 bf16_t;
typedef __attribute__((ext_vector_type(16))) __bf16 v16bf;
typedef __attribute__((ext_vector_type(8)))  float  v8f;
typedef __attribute__((ext_vector_type(4)))  unsigned int v4u;
typedef __attribute__((ext_vector_type(8)))  int  v8i;
typedef __attribute__((ext_vector_type(4)))  int  v4i;

#if __has_builtin(__builtin_amdgcn_tensor_load_to_lds) && \
    __has_builtin(__builtin_amdgcn_s_wait_tensorcnt)
#define HAVE_TDM 1
#else
#define HAVE_TDM 0
#endif

union Frag {
  v16bf        v;
  unsigned int u[8];
  __bf16       e[16];
};

// A-fragment K index for 16-bit 16x32 A matrix (contiguous order, matches B):
// lanes 0-15: V0..7 -> K=0..15 pairs; lanes 16-31: K=16..31.
__device__ __forceinline__ int kbOf(int j, int hi) {
  return 2 * j + 16 * hi;
}

// Load 16 contiguous bf16 (32 bytes, 16B-aligned) into a fragment.
__device__ __forceinline__ void load32B(Frag& f, const bf16_t* p) {
  const uint4* q = (const uint4*)p;
  uint4 a = q[0], b = q[1];
  f.u[0] = a.x; f.u[1] = a.y; f.u[2] = a.z; f.u[3] = a.w;
  f.u[4] = b.x; f.u[5] = b.y; f.u[6] = b.z; f.u[7] = b.w;
}

__device__ __forceinline__ v8f bwmma(const Frag& a, const Frag& b, v8f c) {
  return __builtin_amdgcn_wmma_f32_16x16x32_bf16(
      /*neg_a=*/false, a.v, /*neg_b=*/false, b.v,
      /*c_mod=*/(short)0, c, /*reuse_a=*/false, /*reuse_b=*/false);
}

// ---------------------------------------------------------------------------
// Conversion kernels
// ---------------------------------------------------------------------------
__global__ void k_convert(const float* __restrict__ src, bf16_t* __restrict__ dst, int n) {
  int i = blockIdx.x * blockDim.x + threadIdx.x;
  if (i < n) dst[i] = (bf16_t)src[i];
}

// W [K][N] f32 -> Wt [N][K] bf16 (so B-fragments load contiguously).
__global__ void k_convert_t(const float* __restrict__ w, bf16_t* __restrict__ wt, int K, int N) {
  int i = blockIdx.x * blockDim.x + threadIdx.x;
  if (i < K * N) {
    int k = i / N, n = i % N;
    wt[(size_t)n * K + k] = (bf16_t)w[i];
  }
}

// ---------------------------------------------------------------------------
// WMMA GEMM: C[M,N] = A[M,K] (row-major bf16) @ W, with Wt = W^T [N][K] bf16.
// One block per 16-row A strip; the strip (16 x K bf16) is staged into LDS by
// a single Tensor-Data-Mover descriptor (pad 1 dword / 256 dwords -> padded
// row stride 2056B avoids ds_read bank conflicts). 8 waves x two 16x64 output
// tiles cover N=1024; one A-fragment (8 ds_read_b32) feeds 8 WMMAs per k-step.
// mode 0: C bf16 row-major; mode 1: C bf16 transposed [N][M]; mode 2: C f32.
// ---------------------------------------------------------------------------
__global__ void k_gemm(const bf16_t* __restrict__ A, const bf16_t* __restrict__ Wt,
                       void* __restrict__ Cout, int M, int N, int K, int mode) {
  __shared__ unsigned int AtileU[16 * 514 + 4];   // 16 rows @ 2056B padded stride
  int tm   = blockIdx.x;
  int wv   = threadIdx.x >> 5;
  int lane = threadIdx.x & 31;
  int m = lane & 15, hi = lane >> 4;

  const bf16_t* Ab = A + (size_t)tm * 16 * K;     // contiguous 16 x K strip

#if HAVE_TDM
  if (threadIdx.x < 32) {
    // Generic LDS address: low 32 bits are the workgroup LDS byte offset.
    unsigned lds_base = (unsigned)(unsigned long long)(void*)&AtileU[0];
    unsigned long long ga = (unsigned long long)Ab;
    v4u g0;
    g0[0] = 1u;                                        // count=1 (valid), user desc
    g0[1] = lds_base;                                  // lds_addr (bytes)
    g0[2] = (unsigned)(ga & 0xffffffffu);              // global_addr[31:0]
    g0[3] = (unsigned)((ga >> 32) & 0x01ffffffu)       // global_addr[56:32]
          | (2u << 30);                                // type=2 ("image")
    v8i g1;
    g1[0] = (1 << 16)                                  // data_size=1 -> 2 bytes
          | (1 << 20)                                  // pad_enable
          | (7 << 22);                                 // pad_interval: 256 dwords
                                                       // pad_amount code 0: 1 dword
    g1[1] = (int)(1024u << 16);                        // tensor_dim0 = 1024 (lo16)
    g1[2] = (int)(16u << 16);                          // dim0 hi=0 | tensor_dim1=16
    g1[3] = (int)(1024u << 16);                        // dim1 hi=0 | tile_dim0=1024
    g1[4] = 16;                                        // tile_dim1=16, tile_dim2=0
    g1[5] = (int)(unsigned)K;                          // tensor_dim0_stride lo32
    g1[6] = 0;                                         // stride0 hi | stride1 lo
    g1[7] = 0;
    v4i g2; g2[0] = 0; g2[1] = 0; g2[2] = 0; g2[3] = 0; // <=2D: groups 2/3 zero
    v4i g3 = g2;
    v8i g4;                                             // extra group (clang-23
    g4[0] = 0; g4[1] = 0; g4[2] = 0; g4[3] = 0;         // 6-arg form), zero-filled
    g4[4] = 0; g4[5] = 0; g4[6] = 0; g4[7] = 0;
    __builtin_amdgcn_tensor_load_to_lds(g0, g1, g2, g3, g4, 0);
    __builtin_amdgcn_s_wait_tensorcnt(0);
  }
#else
  {
    // Manual copy replicating the TDM padded layout (1 dword pad / 256 dwords).
    const unsigned int* src = (const unsigned int*)Ab;
    for (int i = threadIdx.x; i < 16 * 512; i += 256) {
      int row = i >> 9;                                // 512 dwords per row
      int kp  = i & 511;
      AtileU[row * 514 + kp + (kp >= 256 ? 1 : 0)] = src[i];
    }
  }
#endif
  __syncthreads();

  v8f acc[8];
#pragma unroll
  for (int t = 0; t < 8; t++)
#pragma unroll
    for (int r = 0; r < 8; r++) acc[t][r] = 0.f;

  for (int kk = 0; kk < K; kk += 32) {
    Frag a;
    int rowbase = m * 514 + (kk >= 512 ? 1 : 0);       // mid-row pad after 512 elems
#pragma unroll
    for (int j = 0; j < 8; j++)
      a.u[j] = AtileU[rowbase + ((kk + kbOf(j, hi)) >> 1)];
#pragma unroll
    for (int tx = 0; tx < 2; tx++) {
      int tn = wv * 2 + tx;
#pragma unroll
      for (int nt = 0; nt < 4; nt++) {
        Frag b;
        load32B(b, Wt + (size_t)(tn * 64 + nt * 16 + m) * K + kk + 16 * hi);
        acc[tx * 4 + nt] = bwmma(a, b, acc[tx * 4 + nt]);
      }
    }
  }

  if (mode == 0) {
    bf16_t* C = (bf16_t*)Cout;
#pragma unroll
    for (int tx = 0; tx < 2; tx++)
#pragma unroll
      for (int nt = 0; nt < 4; nt++)
#pragma unroll
        for (int r = 0; r < 8; r++)
          C[(size_t)(tm * 16 + r + 8 * hi) * N + (wv * 2 + tx) * 64 + nt * 16 + m] =
              (bf16_t)acc[tx * 4 + nt][r];
  } else if (mode == 1) {
    bf16_t* Ct = (bf16_t*)Cout;
#pragma unroll
    for (int tx = 0; tx < 2; tx++)
#pragma unroll
      for (int nt = 0; nt < 4; nt++) {
        union { uint4 q; __bf16 e[8]; } pk;
#pragma unroll
        for (int r = 0; r < 8; r++) pk.e[r] = (bf16_t)acc[tx * 4 + nt][r];
        *(uint4*)(Ct + (size_t)((wv * 2 + tx) * 64 + nt * 16 + m) * M + tm * 16 + 8 * hi) = pk.q;
      }
  } else {
    float* C = (float*)Cout;
#pragma unroll
    for (int tx = 0; tx < 2; tx++)
#pragma unroll
      for (int nt = 0; nt < 4; nt++)
#pragma unroll
        for (int r = 0; r < 8; r++)
          C[(size_t)(tm * 16 + r + 8 * hi) * N + (wv * 2 + tx) * 64 + nt * 16 + m] =
              acc[tx * 4 + nt][r];
  }
}

// ---------------------------------------------------------------------------
// Retrieval: cosine sims of last-token key vs events, serial top-10 per batch.
// ---------------------------------------------------------------------------
__global__ void k_retrieval(const bf16_t* __restrict__ kb, const float* __restrict__ events,
                            int* __restrict__ topk) {
  __shared__ float sims[MEMSZ];
  __shared__ float sred[256];
  int b = blockIdx.x, tid = threadIdx.x;
  const bf16_t* qrow = kb + (size_t)(b * S_ + S_ - 1) * DM_;

  float part = 0.f;
  for (int i = tid; i < DM_; i += 256) { float v = (float)qrow[i]; part += v * v; }
  sred[tid] = part; __syncthreads();
  for (int s = 128; s > 0; s >>= 1) { if (tid < s) sred[tid] += sred[tid + s]; __syncthreads(); }
  float qden = sqrtf(sred[0]) + 1e-8f;

  for (int e = tid; e < MEMSZ; e += 256) {
    const float* ev = events + (size_t)e * DM_;
    float dot = 0.f, nn = 0.f;
    for (int i = 0; i < DM_; i++) { float x = ev[i]; dot += x * (float)qrow[i]; nn += x * x; }
    sims[e] = dot / (qden * (sqrtf(nn) + 1e-8f));
  }
  __syncthreads();
  if (tid == 0) {
    for (int k = 0; k < KT; k++) {
      int best = 0; float bv = -3e38f;
      for (int i = 0; i < MEMSZ; i++) if (sims[i] > bv) { bv = sims[i]; best = i; }
      topk[b * KT + k] = best;
      sims[best] = -3e38f;
    }
  }
}

__global__ void k_gather(const float* __restrict__ events, const int* __restrict__ topk,
                         bf16_t* __restrict__ retr) {
  int i = blockIdx.x * blockDim.x + threadIdx.x;
  if (i >= B_ * KT * DM_) return;
  int d = i % DM_;
  int ki = (i / DM_) % KT;
  int b = i / (KT * DM_);
  retr[i] = (bf16_t)events[(size_t)topk[b * KT + ki] * DM_ + d];
}

// ---------------------------------------------------------------------------
// Online-softmax update over one 32-key block (two 16-col score tiles held as
// D-layout f32: row m = r + 8*hi in VGPR r, col = lane&15). Row reductions via
// shfl_xor masks 1..8 (stay inside each 16-lane half). Probabilities are
// written to per-wave LDS for the D->A layout transpose.
// ---------------------------------------------------------------------------
__device__ __forceinline__ void online_update(float sc0[8], float sc1[8],
                                              float mrow[8], float lrow[8],
                                              v8f o[4], bf16_t* pl, int lane) {
  int m = lane & 15, hi = lane >> 4;
#pragma unroll
  for (int r = 0; r < 8; r++) {
    float tmax = fmaxf(sc0[r], sc1[r]);
#pragma unroll
    for (int off = 1; off < 16; off <<= 1) tmax = fmaxf(tmax, __shfl_xor(tmax, off, 32));
    float nm = fmaxf(mrow[r], tmax);
    float alpha = __expf(mrow[r] - nm);
    float p0 = __expf(sc0[r] - nm);
    float p1 = __expf(sc1[r] - nm);
    float rs = p0 + p1;
#pragma unroll
    for (int off = 1; off < 16; off <<= 1) rs += __shfl_xor(rs, off, 32);
    lrow[r] = lrow[r] * alpha + rs;
    mrow[r] = nm;
#pragma unroll
    for (int nt = 0; nt < 4; nt++) o[nt][r] *= alpha;
    pl[(r + 8 * hi) * 32 + m]      = (bf16_t)p0;
    pl[(r + 8 * hi) * 32 + 16 + m] = (bf16_t)p1;
  }
}

// ---------------------------------------------------------------------------
// Flash attention: one wave per (b, h, 16-query tile). Memory-token block
// first (no alibi/causal, cols >= 10 masked), then causal-pruned 32-key
// blocks. Q*K^T: 4 WMMAs/block; P*V: 4 WMMAs/block into 4 f32 accumulators.
// ---------------------------------------------------------------------------
__global__ void k_flash(const bf16_t* __restrict__ qb, const bf16_t* __restrict__ kb,
                        const bf16_t* __restrict__ vt, const bf16_t* __restrict__ retr,
                        const float* __restrict__ amask, bf16_t* __restrict__ ctx) {
  __shared__ bf16_t plds[4][16][32];          // 1KB per wave, private per wave
  int wvloc = threadIdx.x >> 5;
  int w    = blockIdx.x * 4 + wvloc;
  int lane = threadIdx.x & 31;
  int b   = w >> 11;                          // / (H_ * 128)
  int rem = w & 2047;
  int h   = rem >> 7;
  int qt  = rem & 127;
  int q_base = qt << 4;
  int m = lane & 15, hi = lane >> 4;
  bf16_t* pl = &plds[wvloc][0][0];

  // Preload Q A-fragments for the two 32-wide HD chunks.
  Frag qa[2];
  const unsigned int* qp =
      (const unsigned int*)(qb + (size_t)(b * S_ + q_base + m) * DM_ + h * HD_);
#pragma unroll
  for (int c = 0; c < 2; c++)
#pragma unroll
    for (int j = 0; j < 8; j++) qa[c].u[j] = qp[(32 * c + kbOf(j, hi)) >> 1];

  float slope = exp2f(-0.5f * (float)(h + 1));
  float mrow[8], lrow[8];
  v8f o[4];
#pragma unroll
  for (int r = 0; r < 8; r++) { mrow[r] = -3e38f; lrow[r] = 0.f; }
#pragma unroll
  for (int t = 0; t < 4; t++)
#pragma unroll
    for (int r = 0; r < 8; r++) o[t][r] = 0.f;

  // ---- memory-token block (keys 0..9 = retrieved events; 10..31 masked) ----
  {
    v8f sv;
#pragma unroll
    for (int r = 0; r < 8; r++) sv[r] = 0.f;
    int key = m;                                // tile-0 column index
#pragma unroll
    for (int c = 0; c < 2; c++) {
      Frag kf;
      if (key < KT)
        load32B(kf, retr + (size_t)(b * KT + key) * DM_ + h * HD_ + 32 * c + 16 * hi);
      else
#pragma unroll
        for (int t = 0; t < 16; t++) kf.e[t] = (bf16_t)0.f;
      sv = bwmma(qa[c], kf, sv);
    }
    float sc0[8], sc1[8];
#pragma unroll
    for (int r = 0; r < 8; r++) {
      float sc = 0.125f * sv[r];
      if (key >= KT) sc = NEGBIG;
      sc0[r] = sc; sc1[r] = NEGBIG;             // tile 1 fully masked
    }
    online_update(sc0, sc1, mrow, lrow, o, pl, lane);

    Frag pa;
    const unsigned int* pp = (const unsigned int*)(pl + m * 32);
#pragma unroll
    for (int j = 0; j < 8; j++) pa.u[j] = pp[kbOf(j, hi) >> 1];
#pragma unroll
    for (int nt = 0; nt < 4; nt++) {
      Frag vb;
#pragma unroll
      for (int t = 0; t < 16; t++) {
        int kk = 16 * hi + t;
        float val = (kk < KT)
            ? (float)retr[(size_t)(b * KT + kk) * DM_ + h * HD_ + nt * 16 + m] : 0.f;
        vb.e[t] = (bf16_t)val;
      }
      o[nt] = bwmma(pa, vb, o[nt]);
    }
  }

  // ---- main causal loop over 32-key blocks (skip fully-masked blocks) ----
  int jmax = (q_base + 15) >> 5;
  for (int j = 0; j <= jmax; j++) {
    float sc0[8], sc1[8];
#pragma unroll
    for (int t = 0; t < 2; t++) {
      int key = 32 * j + 16 * t + m;
      v8f sv;
#pragma unroll
      for (int r = 0; r < 8; r++) sv[r] = 0.f;
      const bf16_t* krow = kb + (size_t)(b * S_ + key) * DM_ + h * HD_;
#pragma unroll
      for (int c = 0; c < 2; c++) {
        Frag kf;
        load32B(kf, krow + 32 * c + 16 * hi);
        sv = bwmma(qa[c], kf, sv);
      }
      float kbias = (1.f - amask[b * S_ + key]) * NEGBIG;
      float* dst = t ? sc1 : sc0;
#pragma unroll
      for (int r = 0; r < 8; r++) {
        int qpos = q_base + r + 8 * hi;
        float sc = 0.125f * sv[r] + slope * (float)(qpos - key) + kbias;
        if (key > qpos) sc = NEGBIG;            // strict causal on real cols
        dst[r] = sc;
      }
    }
    online_update(sc0, sc1, mrow, lrow, o, pl, lane);

    Frag pa;
    const unsigned int* pp = (const unsigned int*)(pl + m * 32);
#pragma unroll
    for (int j2 = 0; j2 < 8; j2++) pa.u[j2] = pp[kbOf(j2, hi) >> 1];
#pragma unroll
    for (int nt = 0; nt < 4; nt++) {
      Frag vb;                                   // V^T: contiguous along keys
      load32B(vb, vt + (size_t)(h * HD_ + nt * 16 + m) * (B_ * S_) + b * S_ + 32 * j + 16 * hi);
      o[nt] = bwmma(pa, vb, o[nt]);
    }
  }

  // ---- epilogue: normalize and store context bf16 [b*S+s][h*64+d] ----
#pragma unroll
  for (int nt = 0; nt < 4; nt++)
#pragma unroll
    for (int r = 0; r < 8; r++) {
      float val = o[nt][r] / lrow[r];
      ctx[(size_t)(b * S_ + q_base + r + 8 * hi) * DM_ + h * HD_ + nt * 16 + m] = (bf16_t)val;
    }
}

// ---------------------------------------------------------------------------
extern "C" void kernel_launch(void* const* d_in, const int* in_sizes, int n_in,
                              void* d_out, int out_size, void* d_ws, size_t ws_size,
                              hipStream_t stream) {
  (void)in_sizes; (void)n_in; (void)out_size; (void)ws_size;
  const float* inputs = (const float*)d_in[0];
  const float* amask  = (const float*)d_in[1];
  const float* Wq     = (const float*)d_in[2];
  const float* Wk     = (const float*)d_in[3];
  const float* Wv     = (const float*)d_in[4];
  const float* Wo     = (const float*)d_in[5];
  const float* events = (const float*)d_in[6];
  float* out = (float*)d_out;

  const size_t MS = (size_t)B_ * S_;   // 4096 rows
  char* p = (char*)d_ws;
  auto alloc = [&](size_t bytes) -> void* {
    void* r = (void*)p;
    p += (bytes + 255) & ~(size_t)255;
    return r;
  };
  bf16_t* Xbf  = (bf16_t*)alloc(MS * DM_ * 2);
  bf16_t* Wqt  = (bf16_t*)alloc((size_t)DM_ * DM_ * 2);
  bf16_t* Wkt  = (bf16_t*)alloc((size_t)DM_ * DM_ * 2);
  bf16_t* Wvt  = (bf16_t*)alloc((size_t)DM_ * DM_ * 2);
  bf16_t* Wot  = (bf16_t*)alloc((size_t)DM_ * DM_ * 2);
  bf16_t* qbuf = (bf16_t*)alloc(MS * DM_ * 2);
  bf16_t* kbuf = (bf16_t*)alloc(MS * DM_ * 2);
  bf16_t* vtb  = (bf16_t*)alloc(MS * DM_ * 2);   // V transposed [DM][B*S]
  bf16_t* ctx  = (bf16_t*)alloc(MS * DM_ * 2);
  bf16_t* retr = (bf16_t*)alloc((size_t)B_ * KT * DM_ * 2);
  int*    topk = (int*)alloc((size_t)B_ * KT * 4);

  const int thr = 256;
  int nX = (int)(MS * DM_);
  k_convert<<<(nX + thr - 1) / thr, thr, 0, stream>>>(inputs, Xbf, nX);
  int nW = DM_ * DM_;
  k_convert_t<<<(nW + thr - 1) / thr, thr, 0, stream>>>(Wq, Wqt, DM_, DM_);
  k_convert_t<<<(nW + thr - 1) / thr, thr, 0, stream>>>(Wk, Wkt, DM_, DM_);
  k_convert_t<<<(nW + thr - 1) / thr, thr, 0, stream>>>(Wv, Wvt, DM_, DM_);
  k_convert_t<<<(nW + thr - 1) / thr, thr, 0, stream>>>(Wo, Wot, DM_, DM_);

  int gblocks = (int)(MS / 16);               // one block per 16-row A strip
  k_gemm<<<gblocks, 256, 0, stream>>>(Xbf, Wqt, qbuf, (int)MS, DM_, DM_, 0);
  k_gemm<<<gblocks, 256, 0, stream>>>(Xbf, Wkt, kbuf, (int)MS, DM_, DM_, 0);
  k_gemm<<<gblocks, 256, 0, stream>>>(Xbf, Wvt, vtb,  (int)MS, DM_, DM_, 1);

  k_retrieval<<<B_, 256, 0, stream>>>(kbuf, events, topk);
  int nG = B_ * KT * DM_;
  k_gather<<<(nG + thr - 1) / thr, thr, 0, stream>>>(events, topk, retr);

  k_flash<<<(B_ * H_ * (S_ / 16)) / 4, 128, 0, stream>>>(qbuf, kbuf, vtb, retr, amask, ctx);

  k_gemm<<<gblocks, 256, 0, stream>>>(ctx, Wot, out, (int)MS, DM_, DM_, 2);
}
